// FTGCN_85727547228227
// MI455X (gfx1250) — matrix operand order repaired and checked by reading, not compiled, not run on hardware
//
#include <hip/hip_runtime.h>

// ---------------------------------------------------------------------------
// FTGCN forward for MI455X (gfx1250, wave32).
//   * GEMMs: v_wmma_f32_16x16x32_bf16, f32 accumulate. Weights pre-packed into
//     the WMMA B-fragment layout; staged in LDS (double-buffered, async
//     global->LDS b128 copies) and shared by all 8 waves of a block. A operands
//     are read as f32 and packed to bf16 in-register (v_perm_b32 path),
//     co-executing with WMMA — no bf16 activation copies exist in memory.
//   * Sparse hops: coalesced float4 gathers + global_atomic_add_f32 scatters
//     (the true roofline term: ~10 GB HBM traffic => ~0.4 ms at 23.3 TB/s).
// Workspace requirement: ~520 MB (layout in kernel_launch).
// ---------------------------------------------------------------------------

typedef __attribute__((ext_vector_type(16))) __bf16 v16bf;
typedef __attribute__((ext_vector_type(2)))  __bf16 v2bf;
typedef __attribute__((ext_vector_type(8)))  float  v8f;
typedef __attribute__((ext_vector_type(4)))  int    i32x4;

#define DIN 128
#define HID 256
#define DOUT 64

// ---- optional gfx1250 async global->LDS copy path --------------------------
#if defined(__has_builtin)
#  if __has_builtin(__builtin_amdgcn_global_load_async_to_lds_b128)
#    define FTG_ASYNC_B128 1
#  elif __has_builtin(__builtin_amdgcn_load_to_lds)
#    define FTG_LOAD_TO_LDS 1
#  endif
#  if __has_builtin(__builtin_amdgcn_s_wait_asynccnt)
#    define FTG_WAIT_ASYNC() __builtin_amdgcn_s_wait_asynccnt(0)
#  endif
#  if __has_builtin(__builtin_amdgcn_cvt_pk_bf16_f32)
#    define FTG_CVT_PK 1
#  endif
#  if __has_builtin(__builtin_amdgcn_perm)
#    define FTG_PERM 1
#  endif
#endif
#ifndef FTG_WAIT_ASYNC
#  define FTG_WAIT_ASYNC()
#endif

typedef __attribute__((address_space(1))) i32x4 ftg_gi32x4;   // global int4
typedef __attribute__((address_space(3))) i32x4 ftg_li32x4;   // LDS int4
typedef __attribute__((address_space(3))) void  ftg_lvoid;

// 16-byte global -> LDS copy. AS1 pointers share the generic 64-bit address;
// AS3 pointers are the low 32 bits of the generic shared address (ISA 10.2).
__device__ __forceinline__ void copy16_to_lds(const void* g, void* l) {
#if defined(FTG_ASYNC_B128)
  ftg_gi32x4* gp = (ftg_gi32x4*)(unsigned long long)g;
  ftg_li32x4* lp = (ftg_li32x4*)(unsigned int)(unsigned long long)l;
  __builtin_amdgcn_global_load_async_to_lds_b128(gp, lp, 0, 0);
#elif defined(FTG_LOAD_TO_LDS)
  __builtin_amdgcn_load_to_lds(
      (void*)g, (ftg_lvoid*)(unsigned int)(unsigned long long)l, 16, 0, 0);
#else
  *reinterpret_cast<uint4*>(l) = *reinterpret_cast<const uint4*>(g);
#endif
}

// ---- f32 -> bf16 round-to-nearest-even ------------------------------------
__device__ __forceinline__ unsigned short f2bf(float f) {   // NaN-safe (weights)
  unsigned int u = __float_as_uint(f);
  if ((u & 0x7fffffffu) > 0x7f800000u) return (unsigned short)((u >> 16) | 0x0040u);
  return (unsigned short)((u + 0x7fffu + ((u >> 16) & 1u)) >> 16);
}
// Two f32 -> packed bf16 dword (activations: finite, round-to-nearest).
__device__ __forceinline__ unsigned int pk_bf2(float lo, float hi) {
#if defined(FTG_CVT_PK)
  union { v2bf v; unsigned int u; } c;
  c.v = __builtin_amdgcn_cvt_pk_bf16_f32(lo, hi);
  return c.u;
#elif defined(FTG_PERM)
  // round-to-nearest (ties away), then one v_perm_b32 packs {hi[31:16],lo[31:16]}
  unsigned int ul = __float_as_uint(lo) + 0x8000u;
  unsigned int uh = __float_as_uint(hi) + 0x8000u;
  return __builtin_amdgcn_perm(uh, ul, 0x07060302u);
#else
  unsigned int ul = __float_as_uint(lo), uh = __float_as_uint(hi);
  ul = (ul + 0x7fffu + ((ul >> 16) & 1u)) >> 16;
  uh = (uh + 0x7fffu + ((uh >> 16) & 1u)) & 0xffff0000u;
  return ul | uh;
#endif
}

// ---- A fragment: read 16 f32, convert to the WMMA 16x32 bf16 A layout ------
// This lane holds K[j..j+7] from p[0..7] and K[j+16..j+23] from p[16..23].
__device__ __forceinline__ v16bf load_a_frag_f32(const float* p) {
  float4 x0 = *reinterpret_cast<const float4*>(p);
  float4 x1 = *reinterpret_cast<const float4*>(p + 4);
  float4 y0 = *reinterpret_cast<const float4*>(p + 16);
  float4 y1 = *reinterpret_cast<const float4*>(p + 20);
  union { unsigned int d[8]; v16bf v; } u;
  u.d[0] = pk_bf2(x0.x, x0.y); u.d[1] = pk_bf2(x0.z, x0.w);
  u.d[2] = pk_bf2(x1.x, x1.y); u.d[3] = pk_bf2(x1.z, x1.w);
  u.d[4] = pk_bf2(y0.x, y0.y); u.d[5] = pk_bf2(y0.z, y0.w);
  u.d[6] = pk_bf2(y1.x, y1.y); u.d[7] = pk_bf2(y1.z, y1.w);
  return u.v;
}
// ---- B fragment from LDS (two ds_load_b128) --------------------------------
__device__ __forceinline__ v16bf load_b_lds(const unsigned int* p) {
  union { uint4 q[2]; v16bf v; } u;
  u.q[0] = *reinterpret_cast<const uint4*>(p);
  u.q[1] = *reinterpret_cast<const uint4*>(p + 4);
  return u.v;
}

// ---- cooperative copy of one packed-B K-tile (NT KB) into LDS --------------
template <int NT>
__device__ __forceinline__ void stage_b_tile(unsigned int* __restrict__ lds,
                                             const unsigned int* __restrict__ g, int tid) {
  constexpr int NV = NT * 64;                    // uint4 count
#pragma unroll
  for (int j = 0; j < (NV + 255) / 256; ++j) {
    int idx = tid + j * 256;
    if ((NV % 256 == 0) || (idx < NV)) {
      copy16_to_lds(reinterpret_cast<const uint4*>(g) + idx,
                    reinterpret_cast<uint4*>(lds) + idx);
    }
  }
}

// ---------------------------------------------------------------------------
// Degree / norm kernels
// ---------------------------------------------------------------------------
__global__ void deg_kernel(const int* __restrict__ dst, float* __restrict__ deg, int E) {
  int i = blockIdx.x * blockDim.x + threadIdx.x;
  if (i < E) atomicAdd(&deg[dst[i]], 1.0f);
}
__global__ void dinv_kernel(float* __restrict__ deg, int N) {
  int i = blockIdx.x * blockDim.x + threadIdx.x;
  if (i < N) { float d = deg[i]; deg[i] = (d > 0.0f) ? rsqrtf(d) : 0.0f; }
}
__global__ void norm_kernel(const int* __restrict__ src, const int* __restrict__ dst,
                            const float* __restrict__ dinv, float* __restrict__ norm, int E) {
  int i = blockIdx.x * blockDim.x + threadIdx.x;
  if (i < E) norm[i] = dinv[src[i]] * dinv[dst[i]];
}

// ---------------------------------------------------------------------------
// Pack row-major f32 weight [Kdim,Nc] into WMMA 16x16x32 bf16 B-fragment layout.
// ---------------------------------------------------------------------------
__global__ void pack_b_wmma(const float* __restrict__ B, unsigned int* __restrict__ Bp,
                            int Kdim, int Nc) {
  int NT = Nc >> 4;
  int total = (Kdim >> 5) * NT * 32;
  int tid = blockIdx.x * blockDim.x + threadIdx.x;
  if (tid >= total) return;
  int lane = tid & 31, tile = tid >> 5;
  int kt = tile / NT, nt = tile % NT;
  int col = nt * 16 + (lane & 15);
  int kb  = kt * 32 + ((lane >> 4) ? 8 : 0);
  unsigned int* out = Bp + ((size_t)tile * 32 + lane) * 8;
#pragma unroll
  for (int j = 0; j < 8; ++j) {
    int k = kb + 2 * j + (j >= 4 ? 8 : 0);
    unsigned int lo = f2bf(B[(size_t)k * Nc + col]);
    unsigned int hi = f2bf(B[(size_t)(k + 1) * Nc + col]);
    out[j] = lo | (hi << 16);
  }
}

// ---------------------------------------------------------------------------
// Fused WMMA GEMM over a virtual concat A = [S0|S1|S2] (f32, converted to bf16
// in-register). C[M, NT*16] = A @ packedB + bias, optional ReLU.
// 1 wave = one 16-row tile x NT column tiles; 8 waves/block share LDS-staged B.
// ---------------------------------------------------------------------------
template <int SEGKT, int NSEG, int NT>
__global__ __launch_bounds__(256) void wmma_gemm_fused(
    const float* __restrict__ S0, const float* __restrict__ S1, const float* __restrict__ S2,
    const unsigned int* __restrict__ Bp, const float* __restrict__ bias,
    float* __restrict__ C, int M, int relu) {
  constexpr int KT   = SEGKT * NSEG;      // total K tiles (K = KT*32)
  constexpr int Nc   = NT * 16;
  constexpr int SEGC = SEGKT * 32;        // columns per source segment
  __shared__ alignas(16) unsigned int ldsB[2 * NT * 256];

  const int tid  = threadIdx.x;
  const int wave = tid >> 5, lane = tid & 31;
  int m0 = (blockIdx.x * 8 + wave) * 16;
  const bool active = m0 < M;
  if (!active) m0 = 0;                    // keep all waves in the barriers
  const int row = m0 + (lane & 15);
  const int k0  = (lane >> 4) * 8;        // A-fragment K base for this lane

  const float* segs[3] = {S0, S1, S2};

  v8f acc[NT];
#pragma unroll
  for (int n = 0; n < NT; ++n)
#pragma unroll
    for (int j = 0; j < 8; ++j) acc[n][j] = 0.0f;

  stage_b_tile<NT>(ldsB, Bp, tid);
  FTG_WAIT_ASYNC();
  __syncthreads();

#pragma unroll
  for (int s = 0; s < NSEG; ++s) {
    const float* base = segs[s] + (size_t)row * SEGC + k0;
#pragma unroll
    for (int kl = 0; kl < SEGKT; ++kl) {
      const int ktg = s * SEGKT + kl;
      unsigned int* cur = ldsB + (ktg & 1) * (NT * 256);
      if (ktg + 1 < KT)                   // prefetch next K tile into other half
        stage_b_tile<NT>(ldsB + ((ktg + 1) & 1) * (NT * 256),
                         Bp + (size_t)(ktg + 1) * NT * 256, tid);
      v16bf a = load_a_frag_f32(base + kl * 32);
      if (kl + 1 < SEGKT) __builtin_prefetch(base + (kl + 1) * 32, 0, 0);
#pragma unroll
      for (int nt = 0; nt < NT; ++nt) {
        v16bf b = load_b_lds(cur + nt * 256 + lane * 8);
        acc[nt] = __builtin_amdgcn_wmma_f32_16x16x32_bf16(
            false, a, false, b, (short)0, acc[nt], false, false);
      }
      FTG_WAIT_ASYNC();
      __syncthreads();
    }
  }

  if (active) {
    // D layout: VGPR i -> row m0 + i + 8*(lane>>4), column lane&15 in N tile.
    int mr = m0 + 8 * (lane >> 4);
    int nc = lane & 15;
#pragma unroll
    for (int nt = 0; nt < NT; ++nt) {
      int n = nt * 16 + nc;
      float bv = bias[n];
#pragma unroll
      for (int i = 0; i < 8; ++i) {
        float v = acc[nt][i] + bv;
        if (relu) v = fmaxf(v, 0.0f);
        C[(size_t)(mr + i) * Nc + n] = v;
      }
    }
  }
}

// ---------------------------------------------------------------------------
// att = softmax(scores,row); h0 = x*att (f32)
// ---------------------------------------------------------------------------
__global__ void att_softmax_mul(const float* __restrict__ scores, const float* __restrict__ x,
                                float* __restrict__ h0) {
  __shared__ float red[DIN];
  int r = blockIdx.x, t = threadIdx.x;
  float s = scores[(size_t)r * DIN + t];
  red[t] = s; __syncthreads();
  for (int k = DIN / 2; k > 0; k >>= 1) {
    if (t < k) red[t] = fmaxf(red[t], red[t + k]);
    __syncthreads();
  }
  float m = red[0]; __syncthreads();
  float e = expf(s - m);
  red[t] = e; __syncthreads();
  for (int k = DIN / 2; k > 0; k >>= 1) {
    if (t < k) red[t] += red[t + k];
    __syncthreads();
  }
  h0[(size_t)r * DIN + t] = x[(size_t)r * DIN + t] * (e / red[0]);
}

// ---------------------------------------------------------------------------
// One aggregation hop: out[dst] += norm[e] * h[src].  One wave per edge;
// each lane owns VEC float4 slabs (C = VEC*128 columns).
// ---------------------------------------------------------------------------
template <int VEC>
__global__ void prop_scatter(const float* __restrict__ h, const int* __restrict__ src,
                             const int* __restrict__ dst, const float* __restrict__ norm,
                             float* __restrict__ out, int E) {
  const int C = VEC * 128;
  int e = (blockIdx.x * blockDim.x + threadIdx.x) >> 5;
  int lane = threadIdx.x & 31;
  if (e >= E) return;
  int s = src[e], d = dst[e];
  float w = norm[e];
  const float4* hs = reinterpret_cast<const float4*>(h + (size_t)s * C);
  float* od = out + (size_t)d * C;
#pragma unroll
  for (int v = 0; v < VEC; ++v) {
    int i4 = lane + v * 32;
    float4 val = hs[i4];
    atomicAdd(&od[i4 * 4 + 0], w * val.x);
    atomicAdd(&od[i4 * 4 + 1], w * val.y);
    atomicAdd(&od[i4 * 4 + 2], w * val.z);
    atomicAdd(&od[i4 * 4 + 3], w * val.w);
  }
}

// ---------------------------------------------------------------------------
// In-place row log_softmax over 64 columns.
// ---------------------------------------------------------------------------
__global__ void log_softmax64(float* __restrict__ y) {
  __shared__ float red[DOUT];
  int r = blockIdx.x, t = threadIdx.x;
  float s = y[(size_t)r * DOUT + t];
  red[t] = s; __syncthreads();
  for (int k = DOUT / 2; k > 0; k >>= 1) {
    if (t < k) red[t] = fmaxf(red[t], red[t + k]);
    __syncthreads();
  }
  float m = red[0]; __syncthreads();
  float tm = s - m;
  red[t] = expf(tm); __syncthreads();
  for (int k = DOUT / 2; k > 0; k >>= 1) {
    if (t < k) red[t] += red[t + k];
    __syncthreads();
  }
  y[(size_t)r * DOUT + t] = tm - logf(red[0]);
}

// ---------------------------------------------------------------------------
extern "C" void kernel_launch(void* const* d_in, const int* in_sizes, int n_in,
                              void* d_out, int out_size, void* d_ws, size_t ws_size,
                              hipStream_t stream) {
  const float* x       = (const float*)d_in[0];
  const int*   ei      = (const int*)  d_in[1];
  const float* att_W   = (const float*)d_in[2];
  const float* att_b   = (const float*)d_in[3];
  const float* conv1_W = (const float*)d_in[4];
  const float* conv1_b = (const float*)d_in[5];
  const float* conv2_W = (const float*)d_in[6];
  const float* conv2_b = (const float*)d_in[7];

  const int N = in_sizes[0] / DIN;
  const int E = in_sizes[1] / 2;
  const int* src  = ei;
  const int* dstp = ei + E;

  // ---- workspace layout ----------------------------------------------------
  char* w = (char*)d_ws;
  size_t off = 0;
  auto take = [&](size_t bytes) { char* p = w + off; off = (off + bytes + 255) & ~(size_t)255; return p; };
  float*        deg = (float*)take((size_t)N * 4);              // then holds dinv
  float*        nrm = (float*)take((size_t)E * 4);
  float*        SC  = (float*)take((size_t)N * DIN * 4);        // att scores
  float*        H0  = (float*)take((size_t)N * DIN * 4);        // gated features
  float*        P1  = (float*)take((size_t)N * DIN * 4);
  float*        P2  = (float*)take((size_t)N * DIN * 4);
  float*        H1  = (float*)take((size_t)N * HID * 4);
  float*        Q1  = (float*)take((size_t)N * HID * 4);
  float*        Q2  = (float*)take((size_t)N * HID * 4);
  unsigned int* WpA = (unsigned int*)take((size_t)DIN * DIN * 2);
  unsigned int* Wp1 = (unsigned int*)take((size_t)3 * DIN * HID * 2);
  unsigned int* Wp2 = (unsigned int*)take((size_t)3 * HID * DOUT * 2);
  float*        out = (float*)d_out;

  const int TB  = 256;
  const int gE  = (E + TB - 1) / TB;
  const int gN  = (N + TB - 1) / TB;
  const int gEw = (E * 32 + TB - 1) / TB;            // one wave per edge
  const int gMt = ((N + 15) / 16 + 7) / 8;           // 8 m-tiles per block

  // 1) degree -> dinv -> edge norm
  (void)hipMemsetAsync(deg, 0, (size_t)N * 4, stream);
  deg_kernel<<<gE, TB, 0, stream>>>(dstp, deg, E);
  dinv_kernel<<<gN, TB, 0, stream>>>(deg, N);
  norm_kernel<<<gE, TB, 0, stream>>>(src, dstp, deg, nrm, E);

  // 2) weight packing into WMMA B-fragment layout
  pack_b_wmma<<<((DIN / 32) * (DIN / 16) * 32 + TB - 1) / TB, TB, 0, stream>>>(att_W, WpA, DIN, DIN);
  pack_b_wmma<<<((3 * DIN / 32) * (HID / 16) * 32 + TB - 1) / TB, TB, 0, stream>>>(conv1_W, Wp1, 3 * DIN, HID);
  pack_b_wmma<<<((3 * HID / 32) * (DOUT / 16) * 32 + TB - 1) / TB, TB, 0, stream>>>(conv2_W, Wp2, 3 * HID, DOUT);

  // 3) attention gate: SC = x @ att_W + b ; h0 = x * softmax(SC)
  wmma_gemm_fused<4, 1, 8><<<gMt, TB, 0, stream>>>(x, x, x, WpA, att_b, SC, N, 0);
  att_softmax_mul<<<N, DIN, 0, stream>>>(SC, x, H0);

  // 4) TAGConv1 hops
  (void)hipMemsetAsync(P1, 0, (size_t)N * DIN * 4, stream);
  prop_scatter<1><<<gEw, TB, 0, stream>>>(H0, src, dstp, nrm, P1, E);
  (void)hipMemsetAsync(P2, 0, (size_t)N * DIN * 4, stream);
  prop_scatter<1><<<gEw, TB, 0, stream>>>(P1, src, dstp, nrm, P2, E);

  // 5) H1 = relu([h0|P1|P2] @ W1 + b1)
  wmma_gemm_fused<4, 3, 16><<<gMt, TB, 0, stream>>>(H0, P1, P2, Wp1, conv1_b, H1, N, 1);

  // 6) TAGConv2 hops
  (void)hipMemsetAsync(Q1, 0, (size_t)N * HID * 4, stream);
  prop_scatter<2><<<gEw, TB, 0, stream>>>(H1, src, dstp, nrm, Q1, E);
  (void)hipMemsetAsync(Q2, 0, (size_t)N * HID * 4, stream);
  prop_scatter<2><<<gEw, TB, 0, stream>>>(Q1, src, dstp, nrm, Q2, E);

  // 7) out = log_softmax([H1|Q1|Q2] @ W2 + b2)
  wmma_gemm_fused<8, 3, 4><<<gMt, TB, 0, stream>>>(H1, Q1, Q2, Wp2, conv2_b, out, N, 0);
  log_softmax64<<<N, DOUT, 0, stream>>>(out);

  (void)n_in; (void)out_size; (void)ws_size;
}
